// NTXentLoss_88021059764342
// MI455X (gfx1250) — matrix operand order repaired and checked by reading, not compiled, not run on hardware
//
#include <hip/hip_runtime.h>

typedef __attribute__((ext_vector_type(16))) _Float16 v16h;
typedef __attribute__((ext_vector_type(8)))  _Float16 v8h;
typedef __attribute__((ext_vector_type(8)))  float    v8f;
typedef __attribute__((ext_vector_type(4)))  int      v4i;

#define HALF_N 4096
#define NROWS  8192          // 2N
#define DIM    256
#define NTILES (NROWS / 16)  // 512 column tiles
#define YSPLIT 4             // column-range split across blockIdx.y
#define TPB    (NTILES / YSPLIT)   // 128 column tiles per block

__device__ __forceinline__ float fast_exp(float x) {
    return __builtin_amdgcn_exp2f(x * 1.44269504088896340736f);
}

// async global -> LDS 16-byte copy (gfx1250, tracked by ASYNCcnt)
__device__ __forceinline__ void async_copy_b128(const _Float16* g, _Float16* l) {
    __builtin_amdgcn_global_load_async_to_lds_b128(
        (__attribute__((address_space(1))) v4i*)(g),
        (__attribute__((address_space(3))) v4i*)(l), 0, 0);
}

__device__ __forceinline__ void wait_async_zero() {
    asm volatile("s_wait_asynccnt 0x0" ::: "memory");
}

// ---------------------------------------------------------------------------
// Kernel 1: L2-normalize rows of [z1;z2] -> zn (row-major f16) and znT (K-major)
// ---------------------------------------------------------------------------
__global__ __launch_bounds__(32)
void k_normalize(const float* __restrict__ z1, const float* __restrict__ z2,
                 _Float16* __restrict__ zn, _Float16* __restrict__ znT) {
    const int row  = blockIdx.x;
    const int lane = threadIdx.x;
    const float* src = (row < HALF_N) ? (z1 + (size_t)row * DIM)
                                      : (z2 + (size_t)(row - HALF_N) * DIM);
    float vals[8];
    float ss = 0.f;
#pragma unroll
    for (int j = 0; j < 8; ++j) {
        vals[j] = src[lane + 32 * j];
        ss += vals[j] * vals[j];
    }
#pragma unroll
    for (int m = 16; m >= 1; m >>= 1) ss += __shfl_xor(ss, m, 32);
    const float inv = 1.0f / fmaxf(sqrtf(ss), 1e-8f);
#pragma unroll
    for (int j = 0; j < 8; ++j) {
        const int k = lane + 32 * j;
        const _Float16 h = (_Float16)(vals[j] * inv);
        zn[(size_t)row * DIM + k] = h;
        znT[(size_t)k * NROWS + row] = h;
    }
}

// ---------------------------------------------------------------------------
// Kernel 2: positive-pair logits pos[i] = (zn_i . zn_{i±N}) / T
// ---------------------------------------------------------------------------
__global__ __launch_bounds__(32)
void k_pos(const _Float16* __restrict__ zn, float* __restrict__ pos) {
    const int row  = blockIdx.x;
    const int lane = threadIdx.x;
    const int pair = (row < HALF_N) ? row + HALF_N : row - HALF_N;
    float acc = 0.f;
#pragma unroll
    for (int j = 0; j < 8; ++j) {
        const int k = lane + 32 * j;
        acc += (float)zn[(size_t)row * DIM + k] * (float)zn[(size_t)pair * DIM + k];
    }
#pragma unroll
    for (int m = 16; m >= 1; m >>= 1) acc += __shfl_xor(acc, m, 32);
    if (lane == 0) pos[row] = acc * 2.0f;   // / TEMPERATURE
}

// ---------------------------------------------------------------------------
// Kernel 3: row sums of exp(sim) via WMMA Gram tiles, with the B K-panel
// staged through double-buffered LDS by async global->LDS copies.
// Grid (64, YSPLIT): block = 8 waves, rows bx*128 + wave*16 .. +15,
// column tiles [by*TPB, by*TPB+TPB).
// ---------------------------------------------------------------------------
__global__ __launch_bounds__(256)
void k_simexp(const _Float16* __restrict__ zn, const _Float16* __restrict__ znT,
              float* __restrict__ Spart) {
    constexpr int BUFH = 256 * 16;                  // halves per LDS buffer (8 KB)
    __shared__ __align__(32) _Float16 ldsB[2 * BUFH];

    const int wave = threadIdx.x >> 5;              // 0..7
    const int lane = threadIdx.x & 31;
    const int m0   = blockIdx.x * 128 + wave * 16;  // wave-exclusive 16 rows
    const int rowA = m0 + (lane & 15);
    const int sel  = (lane >> 4) & 1;
    const int jt0  = blockIdx.y * TPB;

    // Preload 8 A fragments (K=0..255). 16-bit A 16x32 ISA layout:
    // lane = M%16 ; K-halves = {sel*8.., 16+sel*8..}
    v16h a[8];
#pragma unroll
    for (int ks = 0; ks < 8; ++ks) {
        const _Float16* base = zn + (size_t)rowA * DIM + ks * 32 + sel * 8;
        const v8h lo = *(const v8h*)(base);
        const v8h hi = *(const v8h*)(base + 16);
        v16h f;
#pragma unroll
        for (int t = 0; t < 8; ++t) { f[t] = lo[t]; f[t + 8] = hi[t]; }
        a[ks] = f;
    }

    // Cooperative async stage of one B panel (256 K-rows x 16 cols f16 = 8 KB):
    // chunk c (0..511): K-row k=c>>1, 16-byte half h=c&1. Thread does c and c+256.
    auto load_tile = [&](int jt, int buf) {
        const int n0 = jt * 16;
        {
            const int c = threadIdx.x, k = c >> 1, h = c & 1;
            async_copy_b128(znT + (size_t)k * NROWS + n0 + h * 8,
                            ldsB + buf * BUFH + k * 16 + h * 8);
        }
        {
            const int c = threadIdx.x + 256, k = c >> 1, h = c & 1;
            async_copy_b128(znT + (size_t)k * NROWS + n0 + h * 8,
                            ldsB + buf * BUFH + k * 16 + h * 8);
        }
    };

    // prologue: stage first tile
    load_tile(jt0, 0);
    wait_async_zero();
    __syncthreads();

    v8f acc = {};
    for (int it = 0; it < TPB; ++it) {
        const int buf = it & 1;
        if (it + 1 < TPB) load_tile(jt0 + it + 1, buf ^ 1);  // overlap prefetch

        const _Float16* bbase = ldsB + buf * BUFH;
        v8f c = {};
#pragma unroll
        for (int ks = 0; ks < 8; ++ks) {
            // B 32x16 frag: lane%16 = N ; K-halves = {sel*8.., 16+sel*8..} strided in the LDS panel
            v16h b;
#pragma unroll
            for (int t = 0; t < 8; ++t) {
                b[t]     = bbase[(size_t)(ks * 32 + sel * 8 + t) * 16 + (lane & 15)];
                b[t + 8] = bbase[(size_t)(ks * 32 + 16 + sel * 8 + t) * 16 + (lane & 15)];
            }
            c = __builtin_amdgcn_wmma_f32_16x16x32_f16(
                    false, a[ks], false, b, (short)0, c, false, false);
        }
#pragma unroll
        for (int r = 0; r < 8; ++r)
            acc[r] += fast_exp(c[r] * 2.0f);    // logits = dot / T = 2*dot

        wait_async_zero();      // prefetched panel landed
        __syncthreads();        // all waves done reading current panel
    }

    // C layout: VGPR r, lanes 0-15 -> row m0+r ; lanes 16-31 -> row m0+r+8.
#pragma unroll
    for (int r = 0; r < 8; ++r) {
        float v = acc[r];
        v += __shfl_xor(v, 1, 32);
        v += __shfl_xor(v, 2, 32);
        v += __shfl_xor(v, 4, 32);
        v += __shfl_xor(v, 8, 32);
        if ((lane & 15) == 0)
            Spart[(size_t)blockIdx.y * NROWS + m0 + r + sel * 8] = v;
    }
}

// ---------------------------------------------------------------------------
// Kernel 4: loss = mean_i [ log(sum_y Spart[y][i] - e^2) - pos_i ]
// ---------------------------------------------------------------------------
__global__ __launch_bounds__(256)
void k_loss(const float* __restrict__ Spart, const float* __restrict__ pos,
            float* __restrict__ out) {
    __shared__ float red[256];
    const int t = threadIdx.x;
    const float e2 = 7.38905609893065f;   // exp(1/T) self-similarity term
    float acc = 0.f;
    for (int i = t; i < NROWS; i += 256) {
        float s = 0.f;
#pragma unroll
        for (int y = 0; y < YSPLIT; ++y) s += Spart[(size_t)y * NROWS + i];
        acc += logf(fmaxf(s - e2, 1e-20f)) - pos[i];
    }
    red[t] = acc;
    __syncthreads();
    for (int s = 128; s > 0; s >>= 1) {
        if (t < s) red[t] += red[t + s];
        __syncthreads();
    }
    if (t == 0) out[0] = red[0] / (float)NROWS;
}

// ---------------------------------------------------------------------------
extern "C" void kernel_launch(void* const* d_in, const int* in_sizes, int n_in,
                              void* d_out, int out_size, void* d_ws, size_t ws_size,
                              hipStream_t stream) {
    (void)in_sizes; (void)n_in; (void)out_size; (void)ws_size;
    const float* z1 = (const float*)d_in[0];
    const float* z2 = (const float*)d_in[1];

    char* ws = (char*)d_ws;
    const size_t znBytes = (size_t)NROWS * DIM * sizeof(_Float16); // 4 MB
    _Float16* zn    = (_Float16*)(ws);
    _Float16* znT   = (_Float16*)(ws + znBytes);
    float*    Spart = (float*)(ws + 2 * znBytes);                       // 4*8192 f32
    float*    pos   = (float*)(ws + 2 * znBytes +
                               (size_t)YSPLIT * NROWS * sizeof(float));

    k_normalize<<<NROWS, 32, 0, stream>>>(z1, z2, zn, znT);
    k_pos      <<<NROWS, 32, 0, stream>>>(zn, pos);
    dim3 g3(NROWS / 128, YSPLIT);
    k_simexp   <<<g3, 256, 0, stream>>>(zn, znT, Spart);
    k_loss     <<<1, 256, 0, stream>>>(Spart, pos, (float*)d_out);
}
